// TranIUNet_61065845014915
// MI455X (gfx1250) — matrix operand order repaired and checked by reading, not compiled, not run on hardware
//
#include <hip/hip_runtime.h>

typedef __bf16 bf16;
typedef __attribute__((ext_vector_type(16))) __bf16 v16bf;
typedef __attribute__((ext_vector_type(8)))  __bf16 v8bf;
typedef __attribute__((ext_vector_type(8)))  float  v8f;
typedef __attribute__((ext_vector_type(4)))  unsigned int u32x4;
typedef __attribute__((ext_vector_type(8)))  int i32x8;
typedef __attribute__((ext_vector_type(4)))  int i32x4;

#define DEV __device__ __forceinline__

#ifndef __has_builtin
#define __has_builtin(x) 0
#endif
#define HAVE_TDM __has_builtin(__builtin_amdgcn_tensor_load_to_lds)

// ---------------- problem constants ----------------
// B=2, C=64, H=W=128, HEADS=8, HD=8, WIN=8
static constexpr int SLOT   = 2097152;     // anderson history slot stride (elements)
static constexpr int BELEMS = 1048576;     // per-batch flattened size

// ---------------- ws layout (bytes) ----------------
static constexpr size_t OFF_PHILAP = 0;                       // 8 MB  f32
static constexpr size_t OFF_XH     = OFF_PHILAP + 8388608;    // 40 MB f32 (5 slots)
static constexpr size_t OFF_FH     = OFF_XH     + 41943040;   // 40 MB f32
static constexpr size_t OFF_XRES   = OFF_FH     + 41943040;   // 8 MB  f32
static constexpr size_t OFF_XN     = OFF_XRES   + 8388608;    // 4 MB  bf16
static constexpr size_t OFF_QB     = OFF_XN     + 4194304;    // 4 MB  bf16
static constexpr size_t OFF_KB     = OFF_QB     + 4194304;    // 4 MB  bf16
static constexpr size_t OFF_VF     = OFF_KB     + 4194304;    // 8 MB  f32
static constexpr size_t OFF_VTW    = OFF_VF     + 8388608;    // 8 MB  bf16 (4096 win x 16 d x 64 t)
static constexpr size_t OFF_AOUT   = OFF_VTW    + 8388608;    // 8 MB  f32
static constexpr size_t OFF_PIN    = OFF_AOUT   + 8388608;    // 4 MB  bf16
static constexpr size_t OFF_X2     = OFF_PIN    + 4194304;    // 8 MB  f32
static constexpr size_t OFF_XN2    = OFF_X2     + 8388608;    // 4 MB  bf16
static constexpr size_t OFF_HB     = OFF_XN2    + 4194304;    // 8 MB  bf16 (32768 x 128)
static constexpr size_t OFF_WQKVT  = OFF_HB     + 8388608;    // 192x64 bf16
static constexpr size_t OFF_WPT    = OFF_WQKVT  + 24576;      // 64x64 bf16
static constexpr size_t OFF_W1T    = OFF_WPT    + 8192;       // 128x64 bf16
static constexpr size_t OFF_W2T    = OFF_W1T    + 16384;      // 64x128 bf16
static constexpr size_t OFF_GRAM   = OFF_W2T    + 16384;      // 2 x 5x5 f32
static constexpr size_t OFF_ALPHA  = OFF_GRAM   + 512;        // 2 x 8 f32

// ---------------- WMMA helpers ----------------
DEV v8f zero8() { v8f z; for (int i = 0; i < 8; ++i) z[i] = 0.0f; return z; }
DEV v16bf zero16() { v16bf z; for (int i = 0; i < 16; ++i) z[i] = (bf16)0.0f; return z; }

DEV v8f wmma_bf16(v16bf a, v16bf b, v8f c) {
  return __builtin_amdgcn_wmma_f32_16x16x32_bf16(false, a, false, b, (short)0, c, false, false);
}

// A fragment (16x32 bf16, row-major source). lane holds row = lane%16,
// K = 16*(lane/16)+0..15  -> two 16B loads.
DEV v16bf load_a_frag(const bf16* A, int lda, int row, int k0, int lane) {
  const bf16* p = A + (size_t)row * lda + k0 + 16 * (lane >> 4);
  v8bf lo = *(const v8bf*)p;
  v8bf hi = *(const v8bf*)(p + 8);
  v16bf r;
  for (int e = 0; e < 8; ++e) { r[e] = lo[e]; r[e + 8] = hi[e]; }
  return r;
}

// B fragment from pre-transposed weights Bt[N][K] (global or LDS-staged).
// lane holds col = lane%16, K = 16*(lane/16)+0..15 -> one contiguous 32B load.
DEV v16bf load_bt_frag(const bf16* Bt, int ldb, int col, int k0, int lane) {
  const bf16* p = Bt + (size_t)col * ldb + k0 + 16 * (lane >> 4);
  return *(const v16bf*)p;
}

// ---------------- TDM: stage bf16 weight panel into LDS ----------------
// Builds a 1-row-tile D# per CDNA5 ISA ch.8 (count=1, type=2/"image",
// data_size=2B, tensor_dim0=tile_dim0=nelem) and issues TENSOR_LOAD_TO_LDS.
DEV void tdm_stage_b16(const void* gsrc, unsigned lds_off, unsigned nelem) {
#if HAVE_TDM
  unsigned long long ga = (unsigned long long)(uintptr_t)gsrc;
  u32x4 g0;
  g0[0] = 1u;                                               // count=1 (valid user D#)
  g0[1] = lds_off;                                          // lds_addr (bytes)
  g0[2] = (unsigned)ga;                                     // global_addr[31:0]
  g0[3] = (unsigned)((ga >> 32) & 0x01FFFFFFu) | (2u << 30);// global_addr[56:32] | type=2
  i32x8 g1;
  g1[0] = 1 << 16;                                          // data_size=1 -> 2-byte elems
  g1[1] = (int)((nelem & 0xFFFFu) << 16);                   // tensor_dim0[15:0]
  g1[2] = (int)((nelem >> 16) | (1u << 16));                // tensor_dim0[31:16] | tensor_dim1=1
  g1[3] = (int)((nelem & 0xFFFFu) << 16);                   // tile_dim0 = nelem
  g1[4] = 1;                                                // tile_dim1 = 1
  g1[5] = (int)nelem;                                       // tensor_dim0_stride[31:0]
  g1[6] = 0;
  g1[7] = 0;
  i32x4 gz; gz[0] = gz[1] = gz[2] = gz[3] = 0;
#if __clang_major__ >= 23
  i32x8 gz8; for (int i = 0; i < 8; ++i) gz8[i] = 0;
  __builtin_amdgcn_tensor_load_to_lds(g0, g1, gz, gz, gz8, 0);
#else
  __builtin_amdgcn_tensor_load_to_lds(g0, g1, gz, gz, 0);
#endif
  __builtin_amdgcn_s_wait_tensorcnt(0);
#else
  (void)gsrc; (void)lds_off; (void)nelem;
#endif
}

// cooperative staging wrapper: TDM from one wave, barrier for the block
DEV void stage_weights(const bf16* Wt, bf16* sW, int nelem) {
#if HAVE_TDM
  if (threadIdx.x == 0)
    tdm_stage_b16(Wt, (unsigned)(uintptr_t)(void*)sW, (unsigned)nelem);
#else
  for (int i = threadIdx.x; i < nelem / 8; i += 256)
    ((v8bf*)sW)[i] = ((const v8bf*)Wt)[i];
#endif
  __syncthreads();
}

// ---------------- prep kernels ----------------
__global__ __launch_bounds__(256) void k_zero32(unsigned* __restrict__ p) {
  p[(size_t)blockIdx.x * 256 + threadIdx.x] = 0u;
}

__global__ __launch_bounds__(256) void k_philap(const float* __restrict__ phi,
                                                const float* __restrict__ lap,
                                                float* __restrict__ out) {
  int e = blockIdx.x * 256 + threadIdx.x;        // token-major element
  int t = e >> 6, c = e & 63;
  int b = t >> 14, hw = t & 16383;
  size_t src = ((size_t)(b * 64 + c) << 14) + hw;
  out[e] = phi[src] + lap[src];
}

__global__ __launch_bounds__(256) void k_wt(const float* __restrict__ Wq, const float* __restrict__ Wkv,
                                            const float* __restrict__ Wp, const float* __restrict__ W1,
                                            const float* __restrict__ W2,
                                            bf16* __restrict__ WqkvT, bf16* __restrict__ WpT,
                                            bf16* __restrict__ W1T, bf16* __restrict__ W2T) {
  int idx = blockIdx.x * 256 + threadIdx.x;      // 32768 threads exactly
  if (idx < 12288) {                              // WqkvT: 192 x 64
    int n = idx >> 6, kk = idx & 63;
    float v = (n < 64) ? Wq[kk * 64 + n] : Wkv[kk * 128 + (n - 64)];
    WqkvT[idx] = (bf16)v;
  } else if (idx < 16384) {                       // WpT: 64 x 64
    int m = idx - 12288; int n = m >> 6, kk = m & 63;
    WpT[m] = (bf16)Wp[kk * 64 + n];
  } else if (idx < 24576) {                       // W1T: 128 x 64
    int m = idx - 16384; int n = m >> 6, kk = m & 63;
    W1T[m] = (bf16)W1[kk * 128 + n];
  } else {                                        // W2T: 64 x 128
    int m = idx - 24576; int n = m >> 7, kk = m & 127;
    W2T[m] = (bf16)W2[kk * 64 + n];
  }
}

// ---------------- LayerNorm (optionally fused residual-add) ----------------
__global__ __launch_bounds__(256) void k_ln(const float* __restrict__ xin,
                                            const float* __restrict__ addin,
                                            float* __restrict__ resout,
                                            const float* __restrict__ g,
                                            const float* __restrict__ be,
                                            bf16* __restrict__ out) {
  int t = blockIdx.x * 256 + threadIdx.x;
  const float* xp = xin + (size_t)t * 64;
  float xv[64];
  if (addin) {
    const float* ap = addin + (size_t)t * 64;
#pragma unroll
    for (int c = 0; c < 64; ++c) xv[c] = xp[c] + ap[c];
  } else {
#pragma unroll
    for (int c = 0; c < 64; ++c) xv[c] = xp[c];
  }
  if (resout) {
    float* rp = resout + (size_t)t * 64;
#pragma unroll
    for (int c = 0; c < 64; ++c) rp[c] = xv[c];
  }
  float s = 0.f, sq = 0.f;
#pragma unroll
  for (int c = 0; c < 64; ++c) { s += xv[c]; sq += xv[c] * xv[c]; }
  float mu = s * (1.0f / 64.0f);
  float var = sq * (1.0f / 64.0f) - mu * mu;
  float rs = rsqrtf(var + 1e-5f);
  bf16* op = out + (size_t)t * 64;
#pragma unroll
  for (int c = 0; c < 64; ++c) op[c] = (bf16)((xv[c] - mu) * rs * g[c] + be[c]);
}

// ---------------- fused QKV GEMM (M=32768, K=64, N=192) ----------------
__global__ __launch_bounds__(256) void k_gemm_qkv(const bf16* __restrict__ xn, const bf16* __restrict__ Wt,
                                                  bf16* __restrict__ qb, bf16* __restrict__ kb,
                                                  float* __restrict__ vf, bf16* __restrict__ vtw) {
  __shared__ bf16 sW[12288];                     // 192x64 bf16, TDM-staged
  stage_weights(Wt, sW, 12288);
  int lane = threadIdx.x & 31;
  int wave = (blockIdx.x * 256 + threadIdx.x) >> 5;
  int row0 = wave * 16;
  int hi = lane >> 4, nc = lane & 15;
  v16bf a0 = load_a_frag(xn, 64, row0 + nc, 0, lane);
  v16bf a1 = load_a_frag(xn, 64, row0 + nc, 32, lane);
  for (int nt = 0; nt < 12; ++nt) {
    int col = nt * 16 + nc;
    v8f acc = wmma_bf16(a0, load_bt_frag(sW, 64, col, 0, lane), zero8());
    acc = wmma_bf16(a1, load_bt_frag(sW, 64, col, 32, lane), acc);
    if (nt < 4) {
#pragma unroll
      for (int r = 0; r < 8; ++r) {
        int t = row0 + 8 * hi + r;
        qb[(size_t)t * 64 + col] = (bf16)acc[r];
      }
    } else if (nt < 8) {
      int kc = col - 64;
#pragma unroll
      for (int r = 0; r < 8; ++r) {
        int t = row0 + 8 * hi + r;
        kb[(size_t)t * 64 + kc] = (bf16)acc[r];
      }
    } else {
      int vc = col - 128;
      int head = vc >> 3, d = vc & 7;
#pragma unroll
      for (int r = 0; r < 8; ++r) {
        int t = row0 + 8 * hi + r;
        vf[(size_t)t * 64 + vc] = acc[r];
        int b = t >> 14, hw = t & 16383, h = hw >> 7, w = hw & 127;
        int win = (((b << 3) + head) << 8) + ((h >> 3) << 4) + (w >> 3);
        int twin = ((h & 7) << 3) + (w & 7);
        vtw[(size_t)win * 1024 + d * 64 + twin] = (bf16)acc[r];
      }
    }
  }
}

// ---------------- window attention (wave handles 32 query rows of one window) --
__global__ __launch_bounds__(256) void k_attn(const bf16* __restrict__ qb, const bf16* __restrict__ kb,
                                              const bf16* __restrict__ vtw, float* __restrict__ aout) {
  __shared__ bf16 sA[4][64 * 64];                 // post-softmax attn, one window per wave-pair
  int lane = threadIdx.x & 31;
  int gw = (blockIdx.x * 256 + threadIdx.x) >> 5;
  int win = gw >> 1, mh = gw & 1;
  int ws = threadIdx.x >> 6;
  int wx = win & 15, wy = (win >> 4) & 15, head = (win >> 8) & 7, bb = win >> 11;
  int tokbase = (bb << 14) + (wy << 10) + (wx << 3);
  int hi = lane >> 4, lm = lane & 15;

  // ---- S = q @ k^T (head-dim 8 zero-padded into K=32) ----
  v16bf qa[2];
#pragma unroll
  for (int tm = 0; tm < 2; ++tm) {
    qa[tm] = zero16();
    if (lane < 16) {
      int rl = 32 * mh + 16 * tm + lm;
      int t = tokbase + ((rl >> 3) << 7) + (rl & 7);
      v8bf q8 = *(const v8bf*)(qb + (size_t)t * 64 + head * 8);
      for (int e = 0; e < 8; ++e) qa[tm][e] = q8[e];
    }
  }
  v8f s[2][4];
  for (int tn = 0; tn < 4; ++tn) {
    v16bf kf = zero16();
    if (lane < 16) {
      int cl = 16 * tn + lm;
      int t = tokbase + ((cl >> 3) << 7) + (cl & 7);
      v8bf k8 = *(const v8bf*)(kb + (size_t)t * 64 + head * 8);
      for (int e = 0; e < 8; ++e) kf[e] = k8[e];
    }
    s[0][tn] = wmma_bf16(qa[0], kf, zero8());
    s[1][tn] = wmma_bf16(qa[1], kf, zero8());
  }

  // ---- softmax over 64 cols per row (in-register + 16-lane shfl reductions) --
  const float SC = 0.35355339059327373f;          // (C/HEADS)^-0.5
#pragma unroll
  for (int tm = 0; tm < 2; ++tm) {
#pragma unroll
    for (int r = 0; r < 8; ++r) {
      float m = -1e30f;
#pragma unroll
      for (int tn = 0; tn < 4; ++tn) { s[tm][tn][r] *= SC; m = fmaxf(m, s[tm][tn][r]); }
      for (int msk = 1; msk < 16; msk <<= 1) m = fmaxf(m, __shfl_xor(m, msk, 16));
      float sum = 0.f;
#pragma unroll
      for (int tn = 0; tn < 4; ++tn) { float p = __expf(s[tm][tn][r] - m); s[tm][tn][r] = p; sum += p; }
      for (int msk = 1; msk < 16; msk <<= 1) sum += __shfl_xor(sum, msk, 16);
      float inv = 1.0f / sum;
      int row = 32 * mh + 16 * tm + 8 * hi + r;
#pragma unroll
      for (int tn = 0; tn < 4; ++tn)
        sA[ws][row * 64 + 16 * tn + lm] = (bf16)(s[tm][tn][r] * inv);
    }
  }
  __syncthreads();

  // ---- out = attn @ v  (A from LDS, B from transposed-window V) ----
#pragma unroll
  for (int tm = 0; tm < 2; ++tm) {
    v8f acc = zero8();
    for (int k0 = 0; k0 < 64; k0 += 32) {
      int arow = 32 * mh + 16 * tm + lm;
      const bf16* ap = &sA[ws][arow * 64 + k0 + 16 * hi];
      v8bf lo = *(const v8bf*)ap;
      v8bf h8 = *(const v8bf*)(ap + 8);
      v16bf af;
      for (int e = 0; e < 8; ++e) { af[e] = lo[e]; af[e + 8] = h8[e]; }
      v16bf bfr = *(const v16bf*)(vtw + (size_t)win * 1024 + lm * 64 + k0 + 16 * hi);
      acc = wmma_bf16(af, bfr, acc);
    }
    if (lm < 8) {
#pragma unroll
      for (int r = 0; r < 8; ++r) {
        int row = 32 * mh + 16 * tm + 8 * hi + r;
        int t = tokbase + ((row >> 3) << 7) + (row & 7);
        aout[(size_t)t * 64 + head * 8 + lm] = acc[r];
      }
    }
  }
}

// ---------------- lepe: depthwise 3x3 conv on V, fused add of attn out -------
__global__ __launch_bounds__(256) void k_lepe(const float* __restrict__ vf, const float* __restrict__ aout,
                                              const float* __restrict__ dww, const float* __restrict__ dwb,
                                              bf16* __restrict__ pin) {
  __shared__ float sw[64 * 9];
  __shared__ float sb[64];
  for (int i = threadIdx.x; i < 576; i += 256) sw[i] = dww[i];
  if (threadIdx.x < 64) sb[threadIdx.x] = dwb[threadIdx.x];
  __syncthreads();
  int t = blockIdx.x * 256 + threadIdx.x;
  int b = t >> 14, hw = t & 16383, h = hw >> 7, w = hw & 127;
  float acc[64];
#pragma unroll
  for (int c = 0; c < 64; ++c) acc[c] = sb[c];
  for (int dy = -1; dy <= 1; ++dy) {
    int hh = h + dy;
    if ((unsigned)hh >= 128u) continue;
    for (int dx = -1; dx <= 1; ++dx) {
      int w2 = w + dx;
      if ((unsigned)w2 >= 128u) continue;
      const float* vp = vf + (size_t)((b << 14) + (hh << 7) + w2) * 64;
      int j = (dy + 1) * 3 + (dx + 1);
#pragma unroll
      for (int c = 0; c < 64; ++c) acc[c] += vp[c] * sw[c * 9 + j];
    }
  }
  const float* ao = aout + (size_t)t * 64;
  bf16* pp = pin + (size_t)t * 64;
#pragma unroll
  for (int c = 0; c < 64; ++c) pp[c] = (bf16)(acc[c] + ao[c]);
}

// ---------------- proj GEMM: x2 = xres + pin @ Wp + bp ----------------
__global__ __launch_bounds__(256) void k_gemm_proj(const bf16* __restrict__ pin, const bf16* __restrict__ Wt,
                                                   const float* __restrict__ xres, const float* __restrict__ bp,
                                                   float* __restrict__ x2) {
  __shared__ bf16 sW[4096];                      // 64x64 bf16, TDM-staged
  stage_weights(Wt, sW, 4096);
  int lane = threadIdx.x & 31;
  int wave = (blockIdx.x * 256 + threadIdx.x) >> 5;
  int row0 = wave * 16;
  int hi = lane >> 4, nc = lane & 15;
  v16bf a0 = load_a_frag(pin, 64, row0 + nc, 0, lane);
  v16bf a1 = load_a_frag(pin, 64, row0 + nc, 32, lane);
  for (int nt = 0; nt < 4; ++nt) {
    int col = nt * 16 + nc;
    v8f acc = wmma_bf16(a0, load_bt_frag(sW, 64, col, 0, lane), zero8());
    acc = wmma_bf16(a1, load_bt_frag(sW, 64, col, 32, lane), acc);
    float bc = bp[col];
#pragma unroll
    for (int r = 0; r < 8; ++r) {
      int t = row0 + 8 * hi + r;
      x2[(size_t)t * 64 + col] = xres[(size_t)t * 64 + col] + acc[r] + bc;
    }
  }
}

// ---------------- MLP GEMM 1: hb = gelu(xn2 @ W1 + b1f), N=128 ----------------
__global__ __launch_bounds__(256) void k_gemm_mlp1(const bf16* __restrict__ xn2, const bf16* __restrict__ Wt,
                                                   const float* __restrict__ b1f, bf16* __restrict__ hb) {
  __shared__ bf16 sW[8192];                      // 128x64 bf16, TDM-staged
  stage_weights(Wt, sW, 8192);
  int lane = threadIdx.x & 31;
  int wave = (blockIdx.x * 256 + threadIdx.x) >> 5;
  int row0 = wave * 16;
  int hi = lane >> 4, nc = lane & 15;
  v16bf a0 = load_a_frag(xn2, 64, row0 + nc, 0, lane);
  v16bf a1 = load_a_frag(xn2, 64, row0 + nc, 32, lane);
  for (int nt = 0; nt < 8; ++nt) {
    int col = nt * 16 + nc;
    v8f acc = wmma_bf16(a0, load_bt_frag(sW, 64, col, 0, lane), zero8());
    acc = wmma_bf16(a1, load_bt_frag(sW, 64, col, 32, lane), acc);
    float bc = b1f[col];
#pragma unroll
    for (int r = 0; r < 8; ++r) {
      int t = row0 + 8 * hi + r;
      float x = acc[r] + bc;
      float g = 0.5f * x * (1.0f + tanhf(0.7978845608028654f * (x + 0.044715f * x * x * x)));
      hb[(size_t)t * 128 + col] = (bf16)g;
    }
  }
}

// ---------------- MLP GEMM 2: fout = x2 + hb @ W2 + b2f, K=128 ----------------
__global__ __launch_bounds__(256) void k_gemm_mlp2(const bf16* __restrict__ hb, const bf16* __restrict__ Wt,
                                                   const float* __restrict__ b2f, const float* __restrict__ x2,
                                                   float* __restrict__ fout) {
  __shared__ bf16 sW[8192];                      // 64x128 bf16, TDM-staged
  stage_weights(Wt, sW, 8192);
  int lane = threadIdx.x & 31;
  int wave = (blockIdx.x * 256 + threadIdx.x) >> 5;
  int row0 = wave * 16;
  int hi = lane >> 4, nc = lane & 15;
  v16bf a[4];
#pragma unroll
  for (int ks = 0; ks < 4; ++ks) a[ks] = load_a_frag(hb, 128, row0 + nc, ks * 32, lane);
  for (int nt = 0; nt < 4; ++nt) {
    int col = nt * 16 + nc;
    v8f acc = zero8();
#pragma unroll
    for (int ks = 0; ks < 4; ++ks)
      acc = wmma_bf16(a[ks], load_bt_frag(sW, 128, col, ks * 32, lane), acc);
    float bc = b2f[col];
#pragma unroll
    for (int r = 0; r < 8; ++r) {
      int t = row0 + 8 * hi + r;
      fout[(size_t)t * 64 + col] = x2[(size_t)t * 64 + col] + acc[r] + bc;
    }
  }
}

// ---------------- Anderson: Gram / solve / mix / output ----------------
__global__ __launch_bounds__(256) void k_gram(const float* __restrict__ xh, const float* __restrict__ fh,
                                              float* __restrict__ gram, int n) {
  __shared__ float red[256];
  int bid = blockIdx.x;
  int b = bid / (n * n);
  int ij = bid % (n * n);
  int i = ij / n, j = ij % n;
  size_t base = (size_t)b * BELEMS;
  const float* fi = fh + (size_t)i * SLOT + base;
  const float* xi = xh + (size_t)i * SLOT + base;
  const float* fj = fh + (size_t)j * SLOT + base;
  const float* xj = xh + (size_t)j * SLOT + base;
  float s = 0.f;
  for (int e = threadIdx.x; e < BELEMS; e += 256)
    s += (fi[e] - xi[e]) * (fj[e] - xj[e]);
  red[threadIdx.x] = s;
  __syncthreads();
  for (int st = 128; st > 0; st >>= 1) {
    if (threadIdx.x < st) red[threadIdx.x] += red[threadIdx.x + st];
    __syncthreads();
  }
  if (threadIdx.x == 0) gram[(b * n + i) * n + j] = red[0];
}

__global__ void k_solve(const float* __restrict__ gram, float* __restrict__ alpha, int n) {
  int b = threadIdx.x;
  if (b >= 2) return;
  int m = n + 1;
  float A[6][7];
  for (int i = 0; i < 6; ++i)
    for (int j = 0; j < 7; ++j) A[i][j] = 0.f;
  for (int j = 1; j < m; ++j) A[0][j] = 1.f;
  for (int i = 1; i < m; ++i) A[i][0] = 1.f;
  for (int i = 1; i < m; ++i)
    for (int j = 1; j < m; ++j)
      A[i][j] = gram[(b * n + (i - 1)) * n + (j - 1)] + ((i == j) ? 1e-4f : 0.f);
  A[0][m] = 1.f;
  for (int c = 0; c < m; ++c) {                    // Gauss-Jordan, partial pivot
    int piv = c;
    float mv = fabsf(A[c][c]);
    for (int r2 = c + 1; r2 < m; ++r2) {
      float v = fabsf(A[r2][c]);
      if (v > mv) { mv = v; piv = r2; }
    }
    if (piv != c)
      for (int j = 0; j <= m; ++j) { float tmp = A[c][j]; A[c][j] = A[piv][j]; A[piv][j] = tmp; }
    float inv = 1.0f / A[c][c];
    for (int j = 0; j <= m; ++j) A[c][j] *= inv;
    for (int r2 = 0; r2 < m; ++r2) {
      if (r2 == c) continue;
      float f = A[r2][c];
      for (int j = 0; j <= m; ++j) A[r2][j] -= f * A[c][j];
    }
  }
  for (int i = 0; i < n; ++i) alpha[b * 8 + i] = A[i + 1][m];
}

__global__ __launch_bounds__(256) void k_mix(const float* __restrict__ fh, const float* __restrict__ alpha,
                                             float* __restrict__ dst, int n) {
  int e = blockIdx.x * 256 + threadIdx.x;
  int b = e >> 20;
  float s = 0.f;
  for (int i = 0; i < n; ++i) s += alpha[b * 8 + i] * fh[(size_t)i * SLOT + e];
  dst[e] = s;   // BETA = 1.0: mix = sum(alpha_i * fh_i)
}

__global__ __launch_bounds__(256) void k_out(const float* __restrict__ src, float* __restrict__ out) {
  int o = blockIdx.x * 256 + threadIdx.x;          // (B,C,H,W) order
  int b = o >> 20, rem = o & 1048575;
  int c = rem >> 14, hw = rem & 16383;
  out[o] = src[(size_t)((b << 14) + hw) * 64 + c];
}

// ---------------- host orchestration ----------------
struct P {
  float *philap, *xres, *vf, *aout, *x2;
  bf16 *xn, *qb, *kb, *vtw, *pin, *xn2, *hb, *WqkvT, *WpT, *W1T, *W2T;
  const float *bp, *dww, *dwb, *g1, *b1, *g2, *b2, *b1f, *b2f;
};

static void run_f(const float* zin, float* fout, const P& p, hipStream_t s) {
  k_ln<<<128, 256, 0, s>>>(zin, p.philap, p.xres, p.g1, p.b1, p.xn);
  k_gemm_qkv<<<256, 256, 0, s>>>(p.xn, p.WqkvT, p.qb, p.kb, p.vf, p.vtw);
  k_attn<<<1024, 256, 0, s>>>(p.qb, p.kb, p.vtw, p.aout);
  k_lepe<<<128, 256, 0, s>>>(p.vf, p.aout, p.dww, p.dwb, p.pin);
  k_gemm_proj<<<256, 256, 0, s>>>(p.pin, p.WpT, p.xres, p.bp, p.x2);
  k_ln<<<128, 256, 0, s>>>(p.x2, nullptr, nullptr, p.g2, p.b2, p.xn2);
  k_gemm_mlp1<<<256, 256, 0, s>>>(p.xn2, p.W1T, p.b1f, p.hb);
  k_gemm_mlp2<<<256, 256, 0, s>>>(p.hb, p.W2T, p.b2f, p.x2, fout);
}

extern "C" void kernel_launch(void* const* d_in, const int* in_sizes, int n_in,
                              void* d_out, int out_size, void* d_ws, size_t ws_size,
                              hipStream_t stream) {
  (void)in_sizes; (void)n_in; (void)out_size; (void)ws_size;
  const float* phi = (const float*)d_in[0];
  const float* lap = (const float*)d_in[1];
  const float* x0  = (const float*)d_in[2];   // zeros
  const float* Wq  = (const float*)d_in[3];
  const float* Wkv = (const float*)d_in[4];
  const float* Wp  = (const float*)d_in[5];
  const float* bp  = (const float*)d_in[6];
  const float* dww = (const float*)d_in[7];
  const float* dwb = (const float*)d_in[8];
  const float* g1  = (const float*)d_in[9];
  const float* b1  = (const float*)d_in[10];
  const float* g2  = (const float*)d_in[11];
  const float* b2  = (const float*)d_in[12];
  const float* W1  = (const float*)d_in[13];
  const float* b1f = (const float*)d_in[14];
  const float* W2  = (const float*)d_in[15];
  const float* b2f = (const float*)d_in[16];

  char* ws = (char*)d_ws;
  float* philap = (float*)(ws + OFF_PHILAP);
  float* xh     = (float*)(ws + OFF_XH);
  float* fh     = (float*)(ws + OFF_FH);
  float* gram   = (float*)(ws + OFF_GRAM);
  float* alpha  = (float*)(ws + OFF_ALPHA);

  P p;
  p.philap = philap;
  p.xres = (float*)(ws + OFF_XRES);
  p.vf   = (float*)(ws + OFF_VF);
  p.aout = (float*)(ws + OFF_AOUT);
  p.x2   = (float*)(ws + OFF_X2);
  p.xn   = (bf16*)(ws + OFF_XN);
  p.qb   = (bf16*)(ws + OFF_QB);
  p.kb   = (bf16*)(ws + OFF_KB);
  p.vtw  = (bf16*)(ws + OFF_VTW);
  p.pin  = (bf16*)(ws + OFF_PIN);
  p.xn2  = (bf16*)(ws + OFF_XN2);
  p.hb   = (bf16*)(ws + OFF_HB);
  p.WqkvT = (bf16*)(ws + OFF_WQKVT);
  p.WpT  = (bf16*)(ws + OFF_WPT);
  p.W1T  = (bf16*)(ws + OFF_W1T);
  p.W2T  = (bf16*)(ws + OFF_W2T);
  p.bp = bp; p.dww = dww; p.dwb = dwb;
  p.g1 = g1; p.b1 = b1; p.g2 = g2; p.b2 = b2;
  p.b1f = b1f; p.b2f = b2f;

  // ---- prep: zero xh slot0 + vtw pad, build philap + bf16 transposed weights
  k_zero32<<<8192, 256, 0, stream>>>((unsigned*)xh);           // 2M words
  k_zero32<<<8192, 256, 0, stream>>>((unsigned*)p.vtw);        // 4M bf16 = 2M words
  k_philap<<<8192, 256, 0, stream>>>(phi, lap, philap);
  k_wt<<<128, 256, 0, stream>>>(Wq, Wkv, Wp, W1, W2, p.WqkvT, p.WpT, p.W1T, p.W2T);

  // ---- Anderson bootstrap: xh0 = 0, fh0 = f(0); xh1 = f0, fh1 = f(f0)
  run_f(x0, fh, p, stream);
  hipMemcpyAsync(xh + (size_t)1 * SLOT, fh, (size_t)SLOT * sizeof(float),
                 hipMemcpyDeviceToDevice, stream);
  run_f(xh + (size_t)1 * SLOT, fh + (size_t)1 * SLOT, p, stream);

  // ---- Anderson iterations k = 2..7 (f eval at k=7 is dead in the reference)
  for (int k = 2; k < 8; ++k) {
    int n = (k < 5) ? k : 5;
    k_gram<<<2 * n * n, 256, 0, stream>>>(xh, fh, gram, n);
    k_solve<<<1, 2, 0, stream>>>(gram, alpha, n);
    float* dst = xh + (size_t)(k % 5) * SLOT;
    k_mix<<<8192, 256, 0, stream>>>(fh, alpha, dst, n);
    if (k < 7)
      run_f(dst, fh + (size_t)(k % 5) * SLOT, p, stream);
  }

  // ---- output: token-major -> (B,C,H,W)
  k_out<<<8192, 256, 0, stream>>>(xh + (size_t)2 * SLOT, (float*)d_out);
}
